// SetPropagation_78426102825591
// MI455X (gfx1250) — matrix operand order repaired and checked by reading, not compiled, not run on hardware
//
#include <hip/hip_runtime.h>

// ---------------------------------------------------------------------------
// Problem constants (from reference setup_inputs)
// ---------------------------------------------------------------------------
#define BB   4
#define N1   2048
#define N2   8192
#define C1   128
#define C2   64
#define CIN  192      // C1 + C2
#define CM   128      // conv output channels
#define NGRP 8        // 128 / 16
#define GSZ  16

typedef __attribute__((ext_vector_type(16))) __bf16          v16bf;
typedef __attribute__((ext_vector_type(8)))  float           v8f;

__device__ __forceinline__ unsigned short f2bf(float f) {
    union { float f; unsigned u; } v; v.f = f;
    unsigned r = v.u + 0x7FFFu + ((v.u >> 16) & 1u);   // round-to-nearest-even
    return (unsigned short)(r >> 16);
}

// ---------------------------------------------------------------------------
// Kernel 1: KNN(3) over LDS-staged targets + inverse-distance weights
// grid (N2/256, B), block 256.  LDS: 3*2048 floats = 24 KB.
// ---------------------------------------------------------------------------
__global__ __launch_bounds__(256)
void knn3_kernel(const float* __restrict__ xyz1, const float* __restrict__ xyz2,
                 int* __restrict__ idxOut, float* __restrict__ wOut)
{
    const int b   = blockIdx.y;
    const int n   = blockIdx.x * 256 + threadIdx.x;
    __shared__ float tx[N1], ty[N1], tz[N1];

    const float* t1 = xyz1 + (size_t)b * 3 * N1;
    for (int i = threadIdx.x; i < N1; i += 256) {
        tx[i] = t1[i];
        ty[i] = t1[N1 + i];
        tz[i] = t1[2 * N1 + i];
    }
    __syncthreads();

    const float qx = xyz2[((size_t)b * 3 + 0) * N2 + n];
    const float qy = xyz2[((size_t)b * 3 + 1) * N2 + n];
    const float qz = xyz2[((size_t)b * 3 + 2) * N2 + n];

    float d0 = 3.4e38f, d1 = 3.4e38f, d2 = 3.4e38f;
    int   i0 = 0, i1 = 0, i2 = 0;
    for (int j = 0; j < N1; ++j) {
        const float dx = qx - tx[j], dy = qy - ty[j], dz = qz - tz[j];
        const float d  = fmaf(dx, dx, fmaf(dy, dy, dz * dz));
        const bool c0 = d < d0, c1 = d < d1, c2 = d < d2;
        i2 = c1 ? i1 : (c2 ? j : i2);  d2 = c1 ? d1 : (c2 ? d : d2);
        i1 = c0 ? i0 : (c1 ? j : i1);  d1 = c0 ? d0 : (c1 ? d : d1);
        i0 = c0 ? j  : i0;             d0 = c0 ? d  : d0;
    }
    const float s0 = sqrtf(d0), s1 = sqrtf(d1), s2 = sqrtf(d2);
    float w0 = 1.0f / (s0 + 1e-8f);
    float w1 = 1.0f / (s1 + 1e-8f);
    float w2 = 1.0f / (s2 + 1e-8f);
    const float inv = 1.0f / (w0 + w1 + w2);
    const size_t base = ((size_t)b * N2 + n) * 3;
    idxOut[base + 0] = i0;  wOut[base + 0] = w0 * inv;
    idxOut[base + 1] = i1;  wOut[base + 1] = w1 * inv;
    idxOut[base + 2] = i2;  wOut[base + 2] = w2 * inv;
}

// ---------------------------------------------------------------------------
// Kernel 2: build X0 = concat(feat2, interp) as bf16, N-major: X0t[b][n][c]
// grid (N2/64, CIN/4, B), block 64.  One ushort4 (4 channels) per thread.
// ---------------------------------------------------------------------------
__global__ __launch_bounds__(64)
void build_xT_kernel(const float* __restrict__ feat1, const float* __restrict__ feat2,
                     const int* __restrict__ idx, const float* __restrict__ w,
                     unsigned short* __restrict__ X0t)
{
    const int n  = blockIdx.x * 64 + threadIdx.x;
    const int c0 = blockIdx.y * 4;
    const int b  = blockIdx.z;

    float v[4];
    if (c0 < C2) {
        #pragma unroll
        for (int i = 0; i < 4; ++i)
            v[i] = feat2[((size_t)b * C2 + c0 + i) * N2 + n];
    } else {
        const size_t ib = ((size_t)b * N2 + n) * 3;
        const int   i0 = idx[ib], i1 = idx[ib + 1], i2 = idx[ib + 2];
        const float w0 = w[ib],  w1 = w[ib + 1],  w2 = w[ib + 2];
        const float* f = feat1 + ((size_t)b * C1 + (c0 - C2)) * N1;
        #pragma unroll
        for (int i = 0; i < 4; ++i) {
            const float* fc = f + (size_t)i * N1;
            v[i] = fmaf(w0, fc[i0], fmaf(w1, fc[i1], w2 * fc[i2]));
        }
    }
    ushort4 o;
    o.x = f2bf(v[0]); o.y = f2bf(v[1]); o.z = f2bf(v[2]); o.w = f2bf(v[3]);
    *(ushort4*)(X0t + ((size_t)b * N2 + n) * CIN + c0) = o;
}

// ---------------------------------------------------------------------------
// Kernel 3: fp32 -> bf16 weight conversion
// ---------------------------------------------------------------------------
__global__ __launch_bounds__(256)
void cvt_w_kernel(const float* __restrict__ w, unsigned short* __restrict__ out, int n)
{
    const int i = blockIdx.x * 256 + threadIdx.x;
    if (i < n) out[i] = f2bf(w[i]);
}

// ---------------------------------------------------------------------------
// Kernel 4: GEMM  Y[b] = Wbf(128xK) * X[b]^T + bias   via v_wmma bf16.
// X stored N-major [B][N2][K] so the 64xK block tile is copied verbatim to
// LDS with async global->LDS b128 copies (ASYNCcnt path), staged ONCE; the
// K-loop is pure ds_load_b128 + v_wmma with no internal barriers.
// block = 256 thr = 8 waves; wave w -> M rows 16w..16w+15; block -> 64 N cols
// grid (N2/64, B)
// ---------------------------------------------------------------------------
template<int K>
__global__ __launch_bounds__(256)
void gemm_bf16_kernel(const unsigned short* __restrict__ Wbf,   // [128 x K]
                      const unsigned short* __restrict__ Xt,    // [B][N2][K]
                      const float* __restrict__ bias,           // [128]
                      float* __restrict__ Y)                    // [B][128][N2]
{
    constexpr int KPAD   = K + 8;        // padded LDS row stride (halfwords)
    constexpr int CHUNKS = 64 * (K / 8); // 16-byte chunks in the 64xK tile

    const int b     = blockIdx.y;
    const int nBase = blockIdx.x * 64;
    const int tid   = threadIdx.x;
    const int wave  = tid >> 5;
    const int lane  = tid & 31;
    const int m0    = wave * 16;
    const int kh    = lane >> 4;    // half-wave select
    const int l16   = lane & 15;

    __shared__ __align__(16) unsigned short ldsX[64 * KPAD];

    // ---- one-shot async copy: 64 rows x K bf16, global [n][K] -> LDS ----
    const unsigned short* Xb = Xt + ((size_t)b * N2 + nBase) * K;
    #pragma unroll
    for (int ch = tid; ch < CHUNKS; ch += 256) {
        const int row = ch / (K / 8);
        const int c8  = ch % (K / 8);
        const unsigned lds_off =
            (unsigned)(size_t)&ldsX[row * KPAD + c8 * 8];
        const unsigned long long gaddr =
            (unsigned long long)(size_t)(Xb + (size_t)row * K + c8 * 8);
        asm volatile("global_load_async_to_lds_b128 %0, %1, off"
                     :: "v"(lds_off), "v"(gaddr) : "memory");
    }
    asm volatile("s_wait_asynccnt 0" ::: "memory");
    __syncthreads();

    v8f acc[4];
    #pragma unroll
    for (int t = 0; t < 4; ++t)
        #pragma unroll
        for (int r = 0; r < 8; ++r) acc[t][r] = 0.0f;

    // A fragment base: 16-bit A 16x32 layout
    // lane<16 : row=l16, elems 0..7 = K k0+0..7,  elems 8..15 = K k0+16..23
    // lane>=16: row=l16, elems 0..7 = K k0+8..15, elems 8..15 = K k0+24..31
    const unsigned short* wrowBase = Wbf + (size_t)(m0 + l16) * K + kh * 8;

    #pragma unroll
    for (int k0 = 0; k0 < K; k0 += 32) {
        union { uint4 q[2]; v16bf v; } au;
        au.q[0] = *(const uint4*)(wrowBase + k0);
        au.q[1] = *(const uint4*)(wrowBase + k0 + 16);
        const v16bf aF = au.v;

        // B fragment: lane = column n (t*16+l16), K order matches A (8h+0..7, 16+8h+0..7)
        #pragma unroll
        for (int t = 0; t < 4; ++t) {
            const unsigned short* brow =
                ldsX + (t * 16 + l16) * KPAD + k0 + kh * 8;
            union { uint4 q[2]; v16bf v; } bu;
            bu.q[0] = *(const uint4*)(brow);
            bu.q[1] = *(const uint4*)(brow + 16);
            acc[t] = __builtin_amdgcn_wmma_f32_16x16x32_bf16(
                         false, aF, false, bu.v, (short)0, acc[t], false, false);
        }
    }

    // ---- epilogue: bias add, store f32 (C/D layout: VGPR r -> M = r+8*kh) ----
    #pragma unroll
    for (int t = 0; t < 4; ++t) {
        const int col = nBase + t * 16 + l16;
        #pragma unroll
        for (int r = 0; r < 8; ++r) {
            const int row = m0 + kh * 8 + r;
            Y[((size_t)b * CM + row) * N2 + col] = acc[t][r] + bias[row];
        }
    }
}

// ---------------------------------------------------------------------------
// Kernel 5: group-norm statistics. grid B*NGRP blocks, 256 thr.
// Deterministic fixed-order tree reduction (no float atomics).
// stats[bg*2] = mean, stats[bg*2+1] = rstd
// ---------------------------------------------------------------------------
__global__ __launch_bounds__(256)
void gn_stats_kernel(const float* __restrict__ Y, float* __restrict__ stats)
{
    const int bg = blockIdx.x;
    const int b  = bg >> 3;
    const int g  = bg & 7;
    const int tid = threadIdx.x;
    __shared__ float ss[256], sq[256];

    float s = 0.0f, q = 0.0f;
    const float* base = Y + ((size_t)b * CM + g * GSZ) * N2;
    for (int c = 0; c < GSZ; ++c)
        for (int n = tid; n < N2; n += 256) {
            const float v = base[(size_t)c * N2 + n];
            s += v; q = fmaf(v, v, q);
        }
    ss[tid] = s; sq[tid] = q;
    __syncthreads();
    for (int off = 128; off > 0; off >>= 1) {
        if (tid < off) { ss[tid] += ss[tid + off]; sq[tid] += sq[tid + off]; }
        __syncthreads();
    }
    if (tid == 0) {
        const float inv = 1.0f / (float)(GSZ * N2);
        const float m   = ss[0] * inv;
        const float var = fmaxf(sq[0] * inv - m * m, 0.0f);
        stats[bg * 2 + 0] = m;
        stats[bg * 2 + 1] = rsqrtf(var + 1e-5f);
    }
}

// ---------------------------------------------------------------------------
// Kernel 6a: normalize + affine + leaky-ReLU -> bf16, N-major X1t[b][n][c]
// grid (N2/64, CM/4, B), block 64.  One ushort4 (4 channels) per thread.
// ---------------------------------------------------------------------------
__global__ __launch_bounds__(64)
void gn_act_bf16T_kernel(const float* __restrict__ Y, const float* __restrict__ stats,
                         const float* __restrict__ gamma, const float* __restrict__ beta,
                         unsigned short* __restrict__ X1t)
{
    const int n  = blockIdx.x * 64 + threadIdx.x;
    const int c0 = blockIdx.y * 4;
    const int b  = blockIdx.z;
    const int g  = c0 >> 4;                       // all 4 channels in one group
    const float m = stats[(b * NGRP + g) * 2 + 0];
    const float r = stats[(b * NGRP + g) * 2 + 1];

    float vv[4];
    #pragma unroll
    for (int i = 0; i < 4; ++i) {
        float v = Y[((size_t)b * CM + c0 + i) * N2 + n];
        v = (v - m) * r * gamma[c0 + i] + beta[c0 + i];
        vv[i] = v >= 0.0f ? v : 0.1f * v;
    }
    ushort4 o;
    o.x = f2bf(vv[0]); o.y = f2bf(vv[1]); o.z = f2bf(vv[2]); o.w = f2bf(vv[3]);
    *(ushort4*)(X1t + ((size_t)b * N2 + n) * CM + c0) = o;
}

// ---------------------------------------------------------------------------
// Kernel 6b: normalize + affine + leaky-ReLU in place on f32 (final output)
// grid (N2/256, 128, B)
// ---------------------------------------------------------------------------
__global__ __launch_bounds__(256)
void gn_act_f32_kernel(float* __restrict__ Y, const float* __restrict__ stats,
                       const float* __restrict__ gamma, const float* __restrict__ beta)
{
    const int n = blockIdx.x * 256 + threadIdx.x;
    const int c = blockIdx.y;
    const int b = blockIdx.z;
    const int g = c >> 4;
    const float m = stats[(b * NGRP + g) * 2 + 0];
    const float r = stats[(b * NGRP + g) * 2 + 1];
    float v = Y[((size_t)b * CM + c) * N2 + n];
    v = (v - m) * r * gamma[c] + beta[c];
    v = v >= 0.0f ? v : 0.1f * v;
    Y[((size_t)b * CM + c) * N2 + n] = v;
}

// ---------------------------------------------------------------------------
// Host launcher
// ---------------------------------------------------------------------------
extern "C" void kernel_launch(void* const* d_in, const int* in_sizes, int n_in,
                              void* d_out, int out_size, void* d_ws, size_t ws_size,
                              hipStream_t stream)
{
    const float* xyz1  = (const float*)d_in[0];
    const float* xyz2  = (const float*)d_in[1];
    const float* feat1 = (const float*)d_in[2];
    const float* feat2 = (const float*)d_in[3];
    const float* w0    = (const float*)d_in[4];
    const float* b0    = (const float*)d_in[5];
    const float* g0    = (const float*)d_in[6];
    const float* beta0 = (const float*)d_in[7];
    const float* w1    = (const float*)d_in[8];
    const float* b1    = (const float*)d_in[9];
    const float* g1    = (const float*)d_in[10];
    const float* beta1 = (const float*)d_in[11];
    float* out = (float*)d_out;

    // ---- workspace layout (bytes) ----
    char* ws = (char*)d_ws;
    int*            idx    = (int*)(ws + 0);                         //  384 KB
    float*          wgt    = (float*)(ws + 393216);                  //  384 KB
    unsigned short* w0bf   = (unsigned short*)(ws + 786432);         //   48 KB
    unsigned short* w1bf   = (unsigned short*)(ws + 835584);         //   32 KB
    float*          stats0 = (float*)(ws + 868352);                  //  256 B
    float*          stats1 = (float*)(ws + 868608);                  //  256 B
    unsigned short* X0t    = (unsigned short*)(ws + 1048576);        // 12.0 MB  [B][N2][192]
    float*          Y0     = (float*)(ws + 1048576 + 12582912);      // 16.0 MB  [B][128][N2]
    unsigned short* X1t    = (unsigned short*)(ws + 1048576 + 12582912 + 16777216); // 8 MB [B][N2][128]
    // total ~38.8 MB

    // 1) KNN + inverse-distance weights
    knn3_kernel<<<dim3(N2 / 256, BB), 256, 0, stream>>>(xyz1, xyz2, idx, wgt);

    // 2) weight conversion to bf16
    cvt_w_kernel<<<(CM * CIN + 255) / 256, 256, 0, stream>>>(w0, w0bf, CM * CIN);
    cvt_w_kernel<<<(CM * CM + 255) / 256, 256, 0, stream>>>(w1, w1bf, CM * CM);

    // 3) X0 = concat(feat2, interp), bf16, N-major
    build_xT_kernel<<<dim3(N2 / 64, CIN / 4, BB), 64, 0, stream>>>(feat1, feat2, idx, wgt, X0t);

    // 4) conv0 GEMM  (WMMA bf16, async global->LDS staging)
    gemm_bf16_kernel<CIN><<<dim3(N2 / 64, BB), 256, 0, stream>>>(w0bf, X0t, b0, Y0);

    // 5) GN0 stats + normalize/lrelu -> X1 (bf16, N-major)
    gn_stats_kernel<<<BB * NGRP, 256, 0, stream>>>(Y0, stats0);
    gn_act_bf16T_kernel<<<dim3(N2 / 64, CM / 4, BB), 64, 0, stream>>>(Y0, stats0, g0, beta0, X1t);

    // 6) conv1 GEMM -> d_out (pre-norm)
    gemm_bf16_kernel<CM><<<dim3(N2 / 64, BB), 256, 0, stream>>>(w1bf, X1t, b1, out);

    // 7) GN1 stats + normalize/lrelu in place on d_out
    gn_stats_kernel<<<BB * NGRP, 256, 0, stream>>>(out, stats1);
    gn_act_f32_kernel<<<dim3(N2 / 256, CM, BB), 256, 0, stream>>>(out, stats1, g1, beta1);
}
